// ConformerLayer_80513456931453
// MI455X (gfx1250) — matrix operand order repaired and checked by reading, not compiled, not run on hardware
//
#include <hip/hip_runtime.h>
#include <math.h>

// ---------------------------------------------------------------------------
// CDNA5 / gfx1250 Conformer layer. All GEMMs use v_wmma_f32_16x16x32_bf16.
// ---------------------------------------------------------------------------

typedef __attribute__((ext_vector_type(16))) __bf16 v16bf;
typedef __attribute__((ext_vector_type(8)))  float  v8f;
typedef __attribute__((ext_vector_type(4)))  unsigned int u32x4;

union Frag {
  u32x4 q[2];
  unsigned short u[16];
  v16bf v;
};

#define DEVI __device__ __forceinline__

DEVI unsigned short f2bf(float f) {
  union { float f; unsigned int u; } v; v.f = f;
  unsigned int r = v.u + 0x7fffu + ((v.u >> 16) & 1u);   // round-nearest-even
  return (unsigned short)(r >> 16);
}
DEVI float bf2f(unsigned short h) {
  union { unsigned int u; float f; } v; v.u = ((unsigned int)h) << 16;
  return v.f;
}
DEVI float sigmoidf_(float x) { return 1.f / (1.f + __expf(-x)); }

DEVI v8f wmma_bf16(const Frag& a, const Frag& b, v8f c) {
  // D = A(16x32 bf16) x B(32x16 bf16) + C(16x16 f32)
  return __builtin_amdgcn_wmma_f32_16x16x32_bf16(
      false, a.v, false, b.v, (short)0, c, false, false);
}

// ---------------------------------------------------------------------------
// Elementwise helpers
// ---------------------------------------------------------------------------
__global__ void k_cvt_bf16(const float* __restrict__ in,
                           unsigned short* __restrict__ out, int n) {
  int i = blockIdx.x * 256 + threadIdx.x;
  if (i < n) out[i] = f2bf(in[i]);
}

// [B,C,T] f32 -> token-major [B*T, C] f32
__global__ void k_tok_major(const float* __restrict__ x, float* __restrict__ xt) {
  int i = blockIdx.x * 256 + threadIdx.x;
  if (i >= 8 * 384 * 1024) return;
  int t = i & 1023;
  int c = (i >> 10) % 384;
  int b = i / (1024 * 384);
  xt[((size_t)(b * 1024 + t)) * 384 + c] = x[i];
}

// ---------------------------------------------------------------------------
// LayerNorm over channels (token-major), bf16 out. One block (128 thr) / token.
// ---------------------------------------------------------------------------
__global__ __launch_bounds__(128)
void k_layernorm(const float* __restrict__ X, const float* __restrict__ g,
                 const float* __restrict__ bb, unsigned short* __restrict__ out) {
  int m = blockIdx.x, tid = threadIdx.x, lane = tid & 31, wv = tid >> 5;
  const float* row = X + (size_t)m * 384;
  float v0 = row[tid], v1 = row[tid + 128], v2 = row[tid + 256];
  float s = v0 + v1 + v2, s2 = v0 * v0 + v1 * v1 + v2 * v2;
  for (int o = 1; o < 32; o <<= 1) { s += __shfl_xor(s, o, 32); s2 += __shfl_xor(s2, o, 32); }
  __shared__ float red[4][2];
  if (lane == 0) { red[wv][0] = s; red[wv][1] = s2; }
  __syncthreads();
  s  = red[0][0] + red[1][0] + red[2][0] + red[3][0];
  s2 = red[0][1] + red[1][1] + red[2][1] + red[3][1];
  float mean = s * (1.f / 384.f);
  float var  = s2 * (1.f / 384.f) - mean * mean;
  float inv  = rsqrtf(var + 1e-5f);
  unsigned short* orow = out + (size_t)m * 384;
  orow[tid]       = f2bf((v0 - mean) * inv * g[tid]       + bb[tid]);
  orow[tid + 128] = f2bf((v1 - mean) * inv * g[tid + 128] + bb[tid + 128]);
  orow[tid + 256] = f2bf((v2 - mean) * inv * g[tid + 256] + bb[tid + 256]);
}

// Final LN + mask, writes [B,C,T] f32
__global__ __launch_bounds__(128)
void k_ln_out(const float* __restrict__ X, const float* __restrict__ g,
              const float* __restrict__ bb, const float* __restrict__ xmask,
              float* __restrict__ out) {
  int m = blockIdx.x, tid = threadIdx.x, lane = tid & 31, wv = tid >> 5;
  const float* row = X + (size_t)m * 384;
  float v0 = row[tid], v1 = row[tid + 128], v2 = row[tid + 256];
  float s = v0 + v1 + v2, s2 = v0 * v0 + v1 * v1 + v2 * v2;
  for (int o = 1; o < 32; o <<= 1) { s += __shfl_xor(s, o, 32); s2 += __shfl_xor(s2, o, 32); }
  __shared__ float red[4][2];
  if (lane == 0) { red[wv][0] = s; red[wv][1] = s2; }
  __syncthreads();
  s  = red[0][0] + red[1][0] + red[2][0] + red[3][0];
  s2 = red[0][1] + red[1][1] + red[2][1] + red[3][1];
  float mean = s * (1.f / 384.f);
  float var  = s2 * (1.f / 384.f) - mean * mean;
  float inv  = rsqrtf(var + 1e-5f);
  int b = m >> 10, t = m & 1023;
  float mk = xmask[m];
  out[((size_t)(b * 384 + tid)) * 1024 + t]       = ((v0 - mean) * inv * g[tid]       + bb[tid])       * mk;
  out[((size_t)(b * 384 + tid + 128)) * 1024 + t] = ((v1 - mean) * inv * g[tid + 128] + bb[tid + 128]) * mk;
  out[((size_t)(b * 384 + tid + 256)) * 1024 + t] = ((v2 - mean) * inv * g[tid + 256] + bb[tid + 256]) * mk;
}

// ---------------------------------------------------------------------------
// Tiled bf16 GEMM: D[M,N] = act(A[M,K] * B[N,K]^T + bias) * scale (+ res)
// 256 threads = 8 waves (fills one WGP), block tile 128x128, wave tile 32x64,
// K-step 32, double-buffered LDS (one barrier per K-step, global loads for the
// next tile overlap the 64 WMMAs of the current one).
// ---------------------------------------------------------------------------
template <bool SILU, bool RES>
__global__ __launch_bounds__(256)
void k_gemm(const unsigned short* __restrict__ A, const unsigned short* __restrict__ B,
            const float* __restrict__ bias, const float* __restrict__ res,
            float* __restrict__ Of, unsigned short* __restrict__ Ob,
            int M, int N, int K, float post_scale, float res_alpha) {
  __shared__ __align__(16) unsigned short sA[2][128 * 32];
  __shared__ __align__(16) unsigned short sB[2][128 * 32];
  const int bm = blockIdx.x * 128, bn = blockIdx.y * 128;
  const int tid = threadIdx.x, lane = tid & 31, wv = tid >> 5;
  const int wm = (wv & 3) * 32, wn = (wv >> 2) * 64;      // 4 x 2 wave grid
  const int half = lane >> 4, l16 = lane & 15;
  v8f acc[2][4] = {};

  // staging: each tile = 128 rows x 32 cols bf16 = 512 u32x4; 256 thr -> 2 each
  u32x4 ra[2], rb[2];
  const int nk = K >> 5;

  auto gload = [&](int k0) {
    #pragma unroll
    for (int i = 0; i < 2; i++) {
      int idx = tid + i * 256;
      int r = idx >> 2, c = idx & 3;
      ra[i] = *(const u32x4*)(A + (size_t)(bm + r) * K + k0 + c * 8);
      rb[i] = *(const u32x4*)(B + (size_t)(bn + r) * K + k0 + c * 8);
    }
  };
  auto sstore = [&](int buf) {
    #pragma unroll
    for (int i = 0; i < 2; i++) {
      int idx = tid + i * 256;
      ((u32x4*)sA[buf])[idx] = ra[i];
      ((u32x4*)sB[buf])[idx] = rb[i];
    }
  };

  gload(0);
  sstore(0);
  __syncthreads();

  for (int ki = 0; ki < nk; ki++) {
    if (ki + 1 < nk) gload((ki + 1) << 5);
    if (ki + 2 < nk) {  // lowers to global_prefetch_b8
      __builtin_prefetch(A + (size_t)(bm + (tid >> 1)) * K + ((ki + 2) << 5), 0, 1);
      __builtin_prefetch(B + (size_t)(bn + (tid >> 1)) * K + ((ki + 2) << 5), 0, 1);
    }
    const unsigned short* cA = sA[ki & 1];
    const unsigned short* cB = sB[ki & 1];
    Frag af[2], bf_[4];
    #pragma unroll
    for (int i = 0; i < 2; i++) {
      // A-frag (16x32): lane<16 holds K {0..7,16..23}, lane>=16 {8..15,24..31}
      const unsigned short* pa = cA + (wm + i * 16 + l16) * 32;
      af[i].q[0] = *(const u32x4*)(pa + half * 8);
      af[i].q[1] = *(const u32x4*)(pa + 16 + half * 8);
    }
    #pragma unroll
    for (int j = 0; j < 4; j++) {
      // B-frag (32x16): lane<16 holds K {0..7,16..23}, lane>=16 {8..15,24..31} (N = l16)
      const unsigned short* pb = cB + (wn + j * 16 + l16) * 32;
      bf_[j].q[0] = *(const u32x4*)(pb + half * 8);
      bf_[j].q[1] = *(const u32x4*)(pb + 16 + half * 8);
    }
    #pragma unroll
    for (int mi = 0; mi < 2; mi++)
      #pragma unroll
      for (int ni = 0; ni < 4; ni++)
        acc[mi][ni] = wmma_bf16(af[mi], bf_[ni], acc[mi][ni]);
    if (ki + 1 < nk) sstore((ki + 1) & 1);
    __syncthreads();
  }

  #pragma unroll
  for (int mi = 0; mi < 2; mi++)
    #pragma unroll
    for (int ni = 0; ni < 4; ni++) {
      int nn = bn + wn + ni * 16 + l16;
      float bs = bias ? bias[nn] : 0.f;
      #pragma unroll
      for (int v = 0; v < 8; v++) {
        int mm = bm + wm + mi * 16 + half * 8 + v;   // C-layout: row = v (+8 hi half)
        float x = acc[mi][ni][v] + bs;
        if (SILU) x = x * sigmoidf_(x);
        x *= post_scale;
        if (RES) x = res[(size_t)mm * N + nn] + res_alpha * x;
        size_t o = (size_t)mm * N + nn;
        if (Of) Of[o] = x;
        if (Ob) Ob[o] = f2bf(x);
      }
    }
}

// ---------------------------------------------------------------------------
// Flash attention core (without rel-v band; exports m,l stats).
// grid (T/64, B*H), block 128 = 4 waves; wave w owns q rows [t0+16w, t0+16w+16)
// ---------------------------------------------------------------------------
__global__ __launch_bounds__(128)
void k_flash(const unsigned short* __restrict__ Q, const unsigned short* __restrict__ Kb,
             const unsigned short* __restrict__ Vb, const float* __restrict__ rk,
             const float* __restrict__ xmask, float* __restrict__ O,
             float* __restrict__ ML) {
  const int T = 1024, C = 384, D = 96;
  const int bh = blockIdx.y, b = bh >> 2, h = bh & 3;
  const int t0 = blockIdx.x * 64;
  const int tid = threadIdx.x, lane = tid & 31, wv = tid >> 5;
  const int half = lane >> 4, l16 = lane & 15;
  __shared__ __align__(16) unsigned short sK[64 * 96];      // [s][d]
  __shared__ __align__(16) unsigned short sV[96 * 64];      // [d][s]  (transposed)
  __shared__ __align__(16) unsigned short sP[4][16 * 64];   // per-wave p tile

  // Q fragments, held in registers for the whole kernel (d=96 -> 3 K-chunks)
  const int trow = t0 + wv * 16 + l16;
  Frag qf[3];
  {
    const unsigned short* qp = Q + (size_t)(b * T + trow) * C + h * D;
    #pragma unroll
    for (int kc = 0; kc < 3; kc++) {
      qf[kc].q[0] = *(const u32x4*)(qp + kc * 32 + half * 8);
      qf[kc].q[1] = *(const u32x4*)(qp + kc * 32 + 16 + half * 8);
    }
  }
  float mrow[8], lrow[8], rmask[8];
  v8f acc[6] = {};
  const int tbase = t0 + wv * 16 + half * 8;
  #pragma unroll
  for (int v = 0; v < 8; v++) {
    mrow[v] = -1e30f; lrow[v] = 0.f;
    rmask[v] = xmask[b * T + tbase + v];
  }

  for (int s0 = 0; s0 < T; s0 += 64) {
    // cooperative stage: K tile row-major, V tile transposed
    #pragma unroll
    for (int i = 0; i < 6; i++) {
      int idx = tid + i * 128;
      int r = idx / 12, c = idx % 12;
      ((u32x4*)sK)[idx] = *(const u32x4*)(Kb + (size_t)(b * T + s0 + r) * C + h * D + c * 8);
      union { u32x4 q; unsigned short u[8]; } vv;
      vv.q = *(const u32x4*)(Vb + (size_t)(b * T + s0 + r) * C + h * D + c * 8);
      #pragma unroll
      for (int j = 0; j < 8; j++) sV[(c * 8 + j) * 64 + r] = vv.u[j];
    }
    __syncthreads();

    // scores: q(16x96) x k^T(96x64) -> 4 n-tiles
    v8f sc[4] = {};
    #pragma unroll
    for (int nt = 0; nt < 4; nt++) {
      #pragma unroll
      for (int kc = 0; kc < 3; kc++) {
        Frag kf;
        const unsigned short* p = sK + (nt * 16 + l16) * 96 + kc * 32;
        kf.q[0] = *(const u32x4*)(p + half * 8);
        kf.q[1] = *(const u32x4*)(p + 16 + half * 8);
        sc[nt] = wmma_bf16(qf[kc], kf, sc[nt]);
      }
    }

    // rel-k band + mask + online softmax (rows live in VGPR index, cols in lane)
    #pragma unroll
    for (int v = 0; v < 8; v++) {
      int t = tbase + v;
      const float* rkrow = rk + ((size_t)bh * T + t) * 9;
      float mx = mrow[v];
      #pragma unroll
      for (int nt = 0; nt < 4; nt++) {
        int s = s0 + nt * 16 + l16;
        float x = sc[nt][v];
        int rel = s - t;
        if (rel >= -4 && rel <= 4) x += rkrow[rel + 4];
        if (rmask[v] * xmask[b * T + s] == 0.f) x = -1e4f;
        sc[nt][v] = x;
        mx = fmaxf(mx, x);
      }
      for (int o = 1; o < 16; o <<= 1) mx = fmaxf(mx, __shfl_xor(mx, o, 32));
      float rescale = __expf(mrow[v] - mx);
      lrow[v] *= rescale;
      #pragma unroll
      for (int j = 0; j < 6; j++) acc[j][v] *= rescale;
      float ss = 0.f;
      #pragma unroll
      for (int nt = 0; nt < 4; nt++) {
        float p = __expf(sc[nt][v] - mx);
        sc[nt][v] = p;
        ss += p;
      }
      for (int o = 1; o < 16; o <<= 1) ss += __shfl_xor(ss, o, 32);
      lrow[v] += ss;
      mrow[v] = mx;
      #pragma unroll
      for (int nt = 0; nt < 4; nt++)   // C-layout -> LDS row-major (for A re-read)
        sP[wv][(half * 8 + v) * 64 + nt * 16 + l16] = f2bf(sc[nt][v]);
    }

    // p(16x64) x V(64x96): A-frags from own-wave sP, B-frags from transposed sV
    Frag pf[2];
    #pragma unroll
    for (int kc = 0; kc < 2; kc++) {
      const unsigned short* p = sP[wv] + l16 * 64 + kc * 32;
      pf[kc].q[0] = *(const u32x4*)(p + half * 8);
      pf[kc].q[1] = *(const u32x4*)(p + 16 + half * 8);
    }
    #pragma unroll
    for (int j = 0; j < 6; j++) {
      #pragma unroll
      for (int kc = 0; kc < 2; kc++) {
        Frag vf;
        const unsigned short* p = sV + (j * 16 + l16) * 64 + kc * 32;
        vf.q[0] = *(const u32x4*)(p + half * 8);
        vf.q[1] = *(const u32x4*)(p + 16 + half * 8);
        acc[j] = wmma_bf16(pf[kc], vf, acc[j]);
      }
    }
    __syncthreads();
  }

  #pragma unroll
  for (int v = 0; v < 8; v++) {
    int t = tbase + v;
    float inv = lrow[v] > 0.f ? 1.f / lrow[v] : 0.f;
    float* orow = O + (size_t)(b * T + t) * C + h * D;
    #pragma unroll
    for (int j = 0; j < 6; j++) orow[j * 16 + l16] = acc[j][v] * inv;
    if (l16 == 0) {
      ML[((size_t)bh * T + t) * 2 + 0] = mrow[v];
      ML[((size_t)bh * T + t) * 2 + 1] = lrow[v];
    }
  }
}

// ---------------------------------------------------------------------------
// Band kernels for relative-position terms (WIN=4 -> 9-wide band)
// ---------------------------------------------------------------------------
__global__ void k_band_qk(const unsigned short* __restrict__ Q,
                          const unsigned short* __restrict__ Kb,
                          const float* __restrict__ relk,
                          float* __restrict__ rk, float* __restrict__ bqk) {
  int i = blockIdx.x * 256 + threadIdx.x;
  const int total = 8 * 4 * 1024 * 9;
  if (i >= total) return;
  int rel = i % 9;
  int t = (i / 9) & 1023;
  int bh = i / (9 * 1024);
  int b = bh >> 2, h = bh & 3;
  const unsigned short* q = Q + (size_t)(b * 1024 + t) * 384 + h * 96;
  int s = t + rel - 4;
  bool sv = (s >= 0 && s < 1024);
  const unsigned short* kp = Kb + (size_t)(b * 1024 + (sv ? s : 0)) * 384 + h * 96;
  float ar = 0.f, ak = 0.f;
  for (int d = 0; d < 96; d++) {
    float qv = bf2f(q[d]);
    ar += qv * relk[rel * 96 + d];
    ak += qv * bf2f(kp[d]);
  }
  rk[i] = ar;
  bqk[i] = sv ? ak : -1e30f;
}

__global__ void k_band_v(const float* __restrict__ bqk, const float* __restrict__ rk,
                         const float* __restrict__ ML, const float* __restrict__ relv,
                         const float* __restrict__ xmask, const float* __restrict__ Oin,
                         unsigned short* __restrict__ Ob) {
  int i = blockIdx.x * 256 + threadIdx.x;
  if (i >= 8192 * 384) return;
  int c = i % 384, mtok = i / 384;
  int b = mtok >> 10, t = mtok & 1023;
  int h = c / 96, d = c % 96;
  size_t row = (size_t)(b * 4 + h) * 1024 + t;
  float mm = ML[row * 2], ll = ML[row * 2 + 1];
  float inv = ll > 0.f ? 1.f / ll : 0.f;
  float o = Oin[i];
  float tm = xmask[mtok];
  for (int rel = 0; rel < 9; rel++) {
    int s = t + rel - 4;
    if (s < 0 || s >= 1024) continue;
    if (tm * xmask[(b << 10) + s] == 0.f) continue;
    float p = __expf(bqk[row * 9 + rel] + rk[row * 9 + rel] - mm) * inv;
    o += p * relv[rel * 96 + d];
  }
  Ob[i] = f2bf(o);
}

// ---------------------------------------------------------------------------
// GLU + depthwise conv (K=7) + BN(eval) + SiLU
// ---------------------------------------------------------------------------
__global__ void k_glu(const float* __restrict__ g1, unsigned short* __restrict__ out) {
  int i = blockIdx.x * 256 + threadIdx.x;
  if (i >= 8192 * 384) return;
  int m = i / 384, c = i % 384;
  float a  = g1[(size_t)m * 768 + c];
  float gt = g1[(size_t)m * 768 + 384 + c];
  out[i] = f2bf(a * sigmoidf_(gt));
}

__global__ void k_dwconv(const unsigned short* __restrict__ x, const float* __restrict__ w,
                         const float* __restrict__ wb, const float* __restrict__ bng,
                         const float* __restrict__ bnb, unsigned short* __restrict__ out) {
  int i = blockIdx.x * 256 + threadIdx.x;
  if (i >= 8192 * 384) return;
  int c = i % 384, mtok = i / 384;
  int b = mtok >> 10, t = mtok & 1023;
  float acc = 0.f;
  #pragma unroll
  for (int j = 0; j < 7; j++) {
    int tt = t + j - 3;
    if (tt < 0 || tt >= 1024) continue;
    acc += bf2f(x[((size_t)((b << 10) + tt)) * 384 + c]) * w[c * 7 + j];
  }
  acc += wb[c];
  acc = acc * rsqrtf(1.f + 1e-5f) * bng[c] + bnb[c];
  acc = acc * sigmoidf_(acc);
  out[i] = f2bf(acc);
}

// ---------------------------------------------------------------------------
// Orchestration
// ---------------------------------------------------------------------------
extern "C" void kernel_launch(void* const* d_in, const int* in_sizes, int n_in,
                              void* d_out, int out_size, void* d_ws, size_t ws_size,
                              hipStream_t stream) {
  (void)in_sizes; (void)n_in; (void)out_size; (void)ws_size;
  const int B = 8, C = 384, T = 1024, M = B * T;

  const float* x        = (const float*)d_in[0];
  const float* xmask    = (const float*)d_in[1];
  const float* ff1_ln_g = (const float*)d_in[2];
  const float* ff1_ln_b = (const float*)d_in[3];
  const float* ff1_w1   = (const float*)d_in[4];
  const float* ff1_b1   = (const float*)d_in[5];
  const float* ff1_w2   = (const float*)d_in[6];
  const float* ff1_b2   = (const float*)d_in[7];
  const float* pre_g    = (const float*)d_in[8];
  const float* pre_b    = (const float*)d_in[9];
  const float* wq = (const float*)d_in[10]; const float* bq = (const float*)d_in[11];
  const float* wk = (const float*)d_in[12]; const float* bk = (const float*)d_in[13];
  const float* wv = (const float*)d_in[14]; const float* bv = (const float*)d_in[15];
  const float* wo = (const float*)d_in[16]; const float* bo = (const float*)d_in[17];
  const float* emb_rel_k = (const float*)d_in[18];
  const float* emb_rel_v = (const float*)d_in[19];
  const float* conv_ln_g = (const float*)d_in[20];
  const float* conv_ln_b = (const float*)d_in[21];
  const float* pw1_w = (const float*)d_in[22]; const float* pw1_b = (const float*)d_in[23];
  const float* dw_w  = (const float*)d_in[24]; const float* dw_b  = (const float*)d_in[25];
  const float* bn_g  = (const float*)d_in[26]; const float* bn_b  = (const float*)d_in[27];
  const float* pw2_w = (const float*)d_in[28]; const float* pw2_b = (const float*)d_in[29];
  const float* ff2_ln_g = (const float*)d_in[30];
  const float* ff2_ln_b = (const float*)d_in[31];
  const float* ff2_w1 = (const float*)d_in[32]; const float* ff2_b1 = (const float*)d_in[33];
  const float* ff2_w2 = (const float*)d_in[34]; const float* ff2_b2 = (const float*)d_in[35];
  const float* post_g = (const float*)d_in[36];
  const float* post_b = (const float*)d_in[37];
  float* out = (float*)d_out;

  // bump allocator on workspace
  char* wsb = (char*)d_ws;
  size_t off = 0;
  auto alloc = [&](size_t bytes) -> void* {
    void* p = wsb + off;
    off += (bytes + 255) & ~(size_t)255;
    return p;
  };
  unsigned short* wb_ff1w1 = (unsigned short*)alloc((size_t)1536 * 384 * 2);
  unsigned short* wb_ff1w2 = (unsigned short*)alloc((size_t)384 * 1536 * 2);
  unsigned short* wb_q  = (unsigned short*)alloc((size_t)384 * 384 * 2);
  unsigned short* wb_k  = (unsigned short*)alloc((size_t)384 * 384 * 2);
  unsigned short* wb_v  = (unsigned short*)alloc((size_t)384 * 384 * 2);
  unsigned short* wb_o  = (unsigned short*)alloc((size_t)384 * 384 * 2);
  unsigned short* wb_p1 = (unsigned short*)alloc((size_t)768 * 384 * 2);
  unsigned short* wb_p2 = (unsigned short*)alloc((size_t)384 * 384 * 2);
  unsigned short* wb_ff2w1 = (unsigned short*)alloc((size_t)1536 * 384 * 2);
  unsigned short* wb_ff2w2 = (unsigned short*)alloc((size_t)384 * 1536 * 2);
  float* r0   = (float*)alloc((size_t)M * C * 4);
  float* r1   = (float*)alloc((size_t)M * C * 4);
  unsigned short* lnb  = (unsigned short*)alloc((size_t)M * C * 2);
  unsigned short* hmid = (unsigned short*)alloc((size_t)M * 1536 * 2);
  float* gbuf = (float*)alloc((size_t)M * 768 * 4);   // GLU input; also attn f32 out
  unsigned short* qb  = (unsigned short*)alloc((size_t)M * C * 2);
  unsigned short* kb  = (unsigned short*)alloc((size_t)M * C * 2);
  unsigned short* vb  = (unsigned short*)alloc((size_t)M * C * 2);
  unsigned short* abf = (unsigned short*)alloc((size_t)M * C * 2);
  float* ml   = (float*)alloc((size_t)B * 4 * T * 2 * 4);
  float* rkb  = (float*)alloc((size_t)B * 4 * T * 9 * 4);
  float* bqkb = (float*)alloc((size_t)B * 4 * T * 9 * 4);
  unsigned short* glub = qb;   // q/k dead after attention -> reuse
  unsigned short* cvb  = kb;

  auto cvt = [&](const float* src, unsigned short* dst, int n) {
    k_cvt_bf16<<<(n + 255) / 256, 256, 0, stream>>>(src, dst, n);
  };
  cvt(ff1_w1, wb_ff1w1, 1536 * 384);
  cvt(ff1_w2, wb_ff1w2, 384 * 1536);
  cvt(wq, wb_q, 384 * 384);  cvt(wk, wb_k, 384 * 384);
  cvt(wv, wb_v, 384 * 384);  cvt(wo, wb_o, 384 * 384);
  cvt(pw1_w, wb_p1, 768 * 384);
  cvt(pw2_w, wb_p2, 384 * 384);
  cvt(ff2_w1, wb_ff2w1, 1536 * 384);
  cvt(ff2_w2, wb_ff2w2, 384 * 1536);

  const int NE = M * C;
  dim3 gblk(256), blk(128);
  auto gemm_grid = [&](int n) { return dim3(M / 128, n / 128); };

  // x -> token major
  k_tok_major<<<(B * C * T + 255) / 256, 256, 0, stream>>>(x, r0);

  // ---- FF1: x1 = x + 0.5 * W2(silu(W1(LN(x)))) ----
  k_layernorm<<<M, blk, 0, stream>>>(r0, ff1_ln_g, ff1_ln_b, lnb);
  k_gemm<true, false><<<gemm_grid(1536), gblk, 0, stream>>>(
      lnb, wb_ff1w1, ff1_b1, nullptr, nullptr, hmid, M, 1536, 384, 1.f, 0.f);
  k_gemm<false, true><<<gemm_grid(384), gblk, 0, stream>>>(
      hmid, wb_ff1w2, ff1_b2, r0, r1, nullptr, M, 384, 1536, 1.f, 0.5f);

  // ---- MHA ----
  k_layernorm<<<M, blk, 0, stream>>>(r1, pre_g, pre_b, lnb);
  const float qscale = 0.1020620726159658f;  // 1/sqrt(96)
  k_gemm<false, false><<<gemm_grid(384), gblk, 0, stream>>>(
      lnb, wb_q, bq, nullptr, nullptr, qb, M, 384, 384, qscale, 0.f);
  k_gemm<false, false><<<gemm_grid(384), gblk, 0, stream>>>(
      lnb, wb_k, bk, nullptr, nullptr, kb, M, 384, 384, 1.f, 0.f);
  k_gemm<false, false><<<gemm_grid(384), gblk, 0, stream>>>(
      lnb, wb_v, bv, nullptr, nullptr, vb, M, 384, 384, 1.f, 0.f);
  k_band_qk<<<(8 * 4 * 1024 * 9 + 255) / 256, 256, 0, stream>>>(qb, kb, emb_rel_k, rkb, bqkb);
  k_flash<<<dim3(T / 64, B * 4), blk, 0, stream>>>(qb, kb, vb, rkb, xmask, gbuf, ml);
  k_band_v<<<(NE + 255) / 256, 256, 0, stream>>>(bqkb, rkb, ml, emb_rel_v, xmask, gbuf, abf);
  k_gemm<false, true><<<gemm_grid(384), gblk, 0, stream>>>(
      abf, wb_o, bo, r1, r0, nullptr, M, 384, 384, 1.f, 1.f);   // x2 in r0

  // ---- Conv block ----
  k_layernorm<<<M, blk, 0, stream>>>(r0, conv_ln_g, conv_ln_b, lnb);
  k_gemm<false, false><<<gemm_grid(768), gblk, 0, stream>>>(
      lnb, wb_p1, pw1_b, nullptr, gbuf, nullptr, M, 768, 384, 1.f, 0.f);
  k_glu<<<(NE + 255) / 256, 256, 0, stream>>>(gbuf, glub);
  k_dwconv<<<(NE + 255) / 256, 256, 0, stream>>>(glub, dw_w, dw_b, bn_g, bn_b, cvb);
  k_gemm<false, true><<<gemm_grid(384), gblk, 0, stream>>>(
      cvb, wb_p2, pw2_b, r0, r1, nullptr, M, 384, 384, 1.f, 1.f);  // x3 in r1

  // ---- FF2 ----
  k_layernorm<<<M, blk, 0, stream>>>(r1, ff2_ln_g, ff2_ln_b, lnb);
  k_gemm<true, false><<<gemm_grid(1536), gblk, 0, stream>>>(
      lnb, wb_ff2w1, ff2_b1, nullptr, nullptr, hmid, M, 1536, 384, 1.f, 0.f);
  k_gemm<false, true><<<gemm_grid(384), gblk, 0, stream>>>(
      hmid, wb_ff2w2, ff2_b2, r1, r0, nullptr, M, 384, 1536, 1.f, 0.5f);  // x4 in r0

  // ---- Final LN + mask -> [B,C,T] ----
  k_ln_out<<<M, blk, 0, stream>>>(r0, post_g, post_b, xmask, out);
}
